// GraphEncoder_15513421873519
// MI455X (gfx1250) — matrix operand order repaired and checked by reading, not compiled, not run on hardware
//
#include <hip/hip_runtime.h>
#include <math.h>
#include <stdint.h>

#define DEV __device__ __forceinline__

typedef __attribute__((ext_vector_type(16))) __bf16 v16bf;
typedef __attribute__((ext_vector_type(8)))  float  v8f;

union Frag16 {
    uint4 q[2];
    v16bf v;
};

DEV unsigned short f2bf(float x) {
    unsigned int u = __float_as_uint(x);
    unsigned int r = u + 0x7FFFu + ((u >> 16) & 1u);   // round-to-nearest-even
    return (unsigned short)(r >> 16);
}

DEV float sigm(float x) { return 1.0f / (1.0f + __expf(-x)); }

static constexpr int NNODES = 16384;   // B*L
static constexpr int BB     = 128;
static constexpr int LL     = 128;
static constexpr int EMBS   = 300;
static constexpr int EMBP   = 320;     // padded to multiple of 32 for K-loop
static constexpr int HHD    = 256;     // per-direction hidden
static constexpr int HD     = 512;
static constexpr int G4     = 1024;    // 4*HHD gates
static constexpr int SS     = 10;

// ---------------------------------------------------------------- embedding
__global__ void k_embed(const int* __restrict__ feat, const float* __restrict__ Wemb,
                        unsigned short* __restrict__ X) {
    int n = blockIdx.x;            // node
    int d = threadIdx.x;           // 0..319
    int tok = feat[n];
    float v = (d < EMBS) ? Wemb[(size_t)tok * EMBS + d] : 0.0f;
    X[(size_t)n * EMBP + d] = f2bf(v);
}

// ------------------------------------------------------- f32 -> bf16 (pad K)
__global__ void k_cast_pad(const float* __restrict__ src, unsigned short* __restrict__ dst,
                           int csrc, int cdst) {
    int r = blockIdx.x;
    for (int c = threadIdx.x; c < cdst; c += blockDim.x) {
        float v = (c < csrc) ? src[(size_t)r * csrc + c] : 0.0f;
        dst[(size_t)r * cdst + c] = f2bf(v);
    }
}

__global__ void k_bias_sum(const float* a, const float* b, float* o, int n) {
    int i = blockIdx.x * blockDim.x + threadIdx.x;
    if (i < n) o[i] = a[i] + b[i];
}

__global__ void k_padrow(const float* __restrict__ pv, float* __restrict__ nodeR) {
    int i = blockIdx.x * blockDim.x + threadIdx.x;
    if (i < HD) nodeR[(size_t)NNODES * HD + i] = pv[i];
}

// ------------------------------------------------------------- tiled GEMM
// C[M,N] = act( A[M,K](bf16, packed lda=K) * W[N,K](bf16)^T + bias[N] )
// One block = one 16-row M tile x 512 output columns.
// The 16xK A strip is staged into LDS once via the CDNA5 async-copy path
// (global_load_async_to_lds_b128, ASYNCcnt), then 8 waves each compute 4
// 16x16 n-tiles, reusing one LDS A fragment across 4 WMMAs per k-step.
__global__ void k_gemm(const unsigned short* __restrict__ A,  // [M,K] bf16
                       const unsigned short* __restrict__ W,  // [N,K] bf16
                       const float* __restrict__ bias,
                       float* __restrict__ C, int ldc,
                       int K, int relu) {
    __shared__ unsigned short As[16 * 1024];                  // up to K=1024
    const int tid  = threadIdx.x;
    const int lane = tid & 31;
    const int wave = tid >> 5;
    const int mt   = blockIdx.x;
    const int ncolbase = blockIdx.y * 512 + wave * 64;

    // ---- async stage A tile (16 x K bf16) into LDS ----
    {
        unsigned ldsbase = (unsigned)(uintptr_t)(&As[0]);
        int rowChunks = K >> 3;                               // uint4 chunks per row
        int chunks    = rowChunks << 4;                       // 16 rows
        for (int i = tid; i < chunks; i += 256) {
            int row = i / rowChunks;
            int col = (i - row * rowChunks) << 3;
            const unsigned short* g = A + (size_t)(mt * 16 + row) * K + col;
            unsigned lds = ldsbase + (unsigned)(row * K + col) * 2u;
            asm volatile("global_load_async_to_lds_b128 %0, %1, off"
                         :: "v"(lds), "v"((unsigned long long)(uintptr_t)g)
                         : "memory");
        }
        asm volatile("s_wait_asynccnt 0x0" ::: "memory");
        __syncthreads();
    }

    const int n  = lane & 15;
    const int hi = lane >> 4;                                 // 0 or 1
    const int ak = hi * 16;
    const int bk = hi * 16;

    v8f acc[4] = {v8f{}, v8f{}, v8f{}, v8f{}};
    const int Kt = K >> 5;
    for (int ks = 0; ks < Kt; ++ks) {
        int kofs = ks * 32;
        Frag16 a;
        a.q[0] = *(const uint4*)&As[n * K + kofs + ak];
        a.q[1] = *(const uint4*)&As[n * K + kofs + 8 + ak];
        for (int j = 0; j < 4; ++j) {
            int ncol = ncolbase + j * 16 + n;
            Frag16 b;
            const unsigned short* wrow = W + (size_t)ncol * K + kofs + bk;
            b.q[0] = *(const uint4*)wrow;
            b.q[1] = *(const uint4*)(wrow + 8);
            acc[j] = __builtin_amdgcn_wmma_f32_16x16x32_bf16(
                false, a.v, false, b.v, (short)0, acc[j], false, false);
        }
    }
    const int mrow = hi * 8;
    for (int j = 0; j < 4; ++j) {
        int ncol = ncolbase + j * 16 + n;
        float bv = bias ? bias[ncol] : 0.0f;
        for (int r = 0; r < 8; ++r) {
            float v = acc[j][r] + bv;
            if (relu) v = fmaxf(v, 0.0f);
            C[(size_t)(mt * 16 + mrow + r) * ldc + ncol] = v;
        }
    }
}

// ---------------------------------------------------------- persistent LSTM
// One block = 16 batch rows, 8 waves. Wave w owns gate columns
// {q*256 + w*32 .. +32} for q = i,f,g,o -> 8 WMMA n-tiles; full recurrence via
// v_wmma_f32_16x16x32_bf16 with h held in LDS as bf16 and c in registers.
__global__ void k_lstm(const float* __restrict__ xp,             // (16384,1024) f32
                       const unsigned short* __restrict__ Whh,   // (1024,256) bf16
                       float* __restrict__ nodeR,                // (16385,512)
                       float* __restrict__ outvec,               // (16384,512)
                       int reverse, int colbase) {
    __shared__ unsigned short hs[16 * HHD];                       // h state, bf16
    const int tid  = threadIdx.x;
    const int lane = tid & 31;
    const int wave = tid >> 5;
    const int b0   = blockIdx.x * 16;

    const int n    = lane & 15;
    const int hi   = lane >> 4;
    const int mrow = hi * 8;

    for (int i = tid; i < 16 * HHD; i += 256) hs[i] = 0;

    float c[2][8];
    for (int j = 0; j < 2; ++j)
        for (int r = 0; r < 8; ++r) c[j][r] = 0.0f;

    for (int step = 0; step < LL; ++step) {
        int t = reverse ? (LL - 1 - step) : step;
        __syncthreads();                                          // h visible

        v8f acc[4][2];
        for (int q = 0; q < 4; ++q)
            for (int j = 0; j < 2; ++j) {
                int ncol = q * 256 + wave * 32 + j * 16 + n;
                v8f a = {};
                for (int r = 0; r < 8; ++r)
                    a[r] = xp[((size_t)(b0 + mrow + r) * LL + t) * G4 + ncol];
                acc[q][j] = a;
            }

        for (int ks = 0; ks < 8; ++ks) {
            int kofs = ks * 32;
            Frag16 a;
            a.q[0] = *(const uint4*)&hs[n * HHD + kofs + hi * 16];
            a.q[1] = *(const uint4*)&hs[n * HHD + kofs + 8 + hi * 16];
            for (int q = 0; q < 4; ++q)
                for (int j = 0; j < 2; ++j) {
                    int ncol = q * 256 + wave * 32 + j * 16 + n;
                    Frag16 b;
                    const unsigned short* wrow = Whh + (size_t)ncol * HHD + kofs + hi * 16;
                    b.q[0] = *(const uint4*)wrow;
                    b.q[1] = *(const uint4*)(wrow + 8);
                    acc[q][j] = __builtin_amdgcn_wmma_f32_16x16x32_bf16(
                        false, a.v, false, b.v, (short)0, acc[q][j], false, false);
                }
        }
        __syncthreads();                                          // reads done

        for (int j = 0; j < 2; ++j)
            for (int r = 0; r < 8; ++r) {
                float iv = sigm(acc[0][j][r]);
                float fv = sigm(acc[1][j][r]);
                float gv = tanhf(acc[2][j][r]);
                float ov = sigm(acc[3][j][r]);
                float cc = fv * c[j][r] + iv * gv;
                c[j][r] = cc;
                float hh = ov * tanhf(cc);
                int hcol = wave * 32 + j * 16 + n;
                hs[(mrow + r) * HHD + hcol] = f2bf(hh);
                size_t node = (size_t)(b0 + mrow + r) * LL + t;
                nodeR [node * HD + colbase + hcol] = hh;
                outvec[node * HD + colbase + hcol] = hh;
            }
    }
}

// ----------------------------------------------- GraphSAGE gather/mean/concat
__global__ void k_gather(const float* __restrict__ selfSrc,   // rows x 512 f32
                         const float* __restrict__ neighSrc,  // rows x 512 f32
                         int padIsZero,                       // idx==N -> zeros
                         const int* __restrict__ adj,         // (N+1,10), rows 0..N-1
                         float* __restrict__ lenBuf, int computeLen,
                         unsigned short* __restrict__ concat) // (N,1024) bf16
{
    int nd  = blockIdx.x;
    int tid = threadIdx.x;            // 256 threads, 2 cols each
    int c0  = tid * 2;
    __shared__ int s_any;
    float s0 = 0.0f, s1 = 0.0f;
    int cnt = 0;
    for (int s = 0; s < SS; ++s) {
        int idx = adj[nd * SS + s];
        float v0 = 0.0f, v1 = 0.0f;
        if (!(padIsZero && idx == NNODES)) {
            v0 = neighSrc[(size_t)idx * HD + c0];
            v1 = neighSrc[(size_t)idx * HD + c0 + 1];
        }
        s0 += v0; s1 += v1;
        if (computeLen) {
            if (tid == 0) s_any = 0;
            __syncthreads();
            if (v0 > 0.0f || v1 > 0.0f) atomicOr(&s_any, 1);
            __syncthreads();
            cnt += s_any;
            __syncthreads();
        }
    }
    float len;
    if (computeLen) {
        len = (float)cnt;
        if (tid == 0) lenBuf[nd] = len;
    } else {
        len = lenBuf[nd];
    }
    float inv = 1.0f / len;
    float a0 = selfSrc[(size_t)nd * HD + c0];
    float a1 = selfSrc[(size_t)nd * HD + c0 + 1];
    size_t base = (size_t)nd * G4;
    concat[base + c0]            = f2bf(a0);
    concat[base + c0 + 1]        = f2bf(a1);
    concat[base + HD + c0]       = f2bf(s0 * inv);
    concat[base + HD + c0 + 1]   = f2bf(s1 * inv);
}

// ----------------------------------------------------- concat + max-pool out
__global__ void k_pool(const float* __restrict__ fwh, const float* __restrict__ bwh,
                       float* __restrict__ hiddenOut, float* __restrict__ gemb) {
    int b = blockIdx.x;
    for (int cc = threadIdx.x; cc < 1024; cc += blockDim.x) {
        const float* src = (cc < HD) ? fwh : bwh;
        int col = (cc < HD) ? cc : cc - HD;
        float mx = -INFINITY;
        for (int l = 0; l < LL; ++l) {
            float v = src[((size_t)b * LL + l) * HD + col];
            hiddenOut[((size_t)b * LL + l) * 1024 + cc] = v;
            mx = fmaxf(mx, v);
        }
        gemb[(size_t)b * 1024 + cc] = mx;                 // == (2B,512) flat
    }
}

// ----------------------------------------------------------------- launcher
extern "C" void kernel_launch(void* const* d_in, const int* in_sizes, int n_in,
                              void* d_out, int out_size, void* d_ws, size_t ws_size,
                              hipStream_t stream) {
    const int*   feature_info = (const int*)d_in[0];
    const int*   fw_adj       = (const int*)d_in[2];
    const int*   bw_adj       = (const int*)d_in[3];
    const float* W_emb        = (const float*)d_in[4];
    const float* Wih_f        = (const float*)d_in[5];
    const float* Whh_f        = (const float*)d_in[6];
    const float* bih_f        = (const float*)d_in[7];
    const float* bhh_f        = (const float*)d_in[8];
    const float* Wih_b        = (const float*)d_in[9];
    const float* Whh_b        = (const float*)d_in[10];
    const float* bih_b        = (const float*)d_in[11];
    const float* bhh_b        = (const float*)d_in[12];
    const float* padding_vec  = (const float*)d_in[13];
    const float* fw_W         = (const float*)d_in[14];
    const float* fw_b         = (const float*)d_in[15];
    const float* bw_W         = (const float*)d_in[16];
    const float* bw_b         = (const float*)d_in[17];

    char* base = (char*)d_ws;
    size_t off = 0;
    auto alloc = [&](size_t bytes) {
        char* p = base + off;
        off = (off + bytes + 255) & ~(size_t)255;
        return p;
    };
    unsigned short* embX   = (unsigned short*)alloc((size_t)NNODES * EMBP * 2);
    unsigned short* wihF   = (unsigned short*)alloc((size_t)G4 * EMBP * 2);
    unsigned short* wihB   = (unsigned short*)alloc((size_t)G4 * EMBP * 2);
    unsigned short* whhF   = (unsigned short*)alloc((size_t)G4 * HHD * 2);
    unsigned short* whhB   = (unsigned short*)alloc((size_t)G4 * HHD * 2);
    unsigned short* sageWF = (unsigned short*)alloc((size_t)7 * HD * G4 * 2);
    unsigned short* sageWB = (unsigned short*)alloc((size_t)7 * HD * G4 * 2);
    float* biasF   = (float*)alloc((size_t)G4 * 4);
    float* biasB   = (float*)alloc((size_t)G4 * 4);
    float* xpF     = (float*)alloc((size_t)NNODES * G4 * 4);
    float* xpB     = (float*)alloc((size_t)NNODES * G4 * 4);
    float* nodeR   = (float*)alloc((size_t)(NNODES + 1) * HD * 4);
    float* hidF    = (float*)alloc((size_t)NNODES * HD * 4);
    float* hidB    = (float*)alloc((size_t)NNODES * HD * 4);
    unsigned short* concatB = (unsigned short*)alloc((size_t)NNODES * G4 * 2);
    float* lenF    = (float*)alloc((size_t)NNODES * 4);
    float* lenB    = (float*)alloc((size_t)NNODES * 4);

    float* out_hidden = (float*)d_out;                        // (B,L,1024)
    float* out_gemb   = out_hidden + (size_t)BB * LL * 1024;  // (2B,512)
    float* out_vec    = out_gemb + (size_t)2 * BB * HD;       // (B,L,512)

    // weight casts
    k_cast_pad<<<dim3(G4), dim3(256), 0, stream>>>(Wih_f, wihF, EMBS, EMBP);
    k_cast_pad<<<dim3(G4), dim3(256), 0, stream>>>(Wih_b, wihB, EMBS, EMBP);
    k_cast_pad<<<dim3(G4), dim3(256), 0, stream>>>(Whh_f, whhF, HHD, HHD);
    k_cast_pad<<<dim3(G4), dim3(256), 0, stream>>>(Whh_b, whhB, HHD, HHD);
    k_cast_pad<<<dim3(7 * HD), dim3(256), 0, stream>>>(fw_W, sageWF, G4, G4);
    k_cast_pad<<<dim3(7 * HD), dim3(256), 0, stream>>>(bw_W, sageWB, G4, G4);
    k_bias_sum<<<dim3(4), dim3(256), 0, stream>>>(bih_f, bhh_f, biasF, G4);
    k_bias_sum<<<dim3(4), dim3(256), 0, stream>>>(bih_b, bhh_b, biasB, G4);

    // embedding gather -> bf16 padded
    k_embed<<<dim3(NNODES), dim3(EMBP), 0, stream>>>(feature_info, W_emb, embX);

    // xp = emb @ Wih^T + (bih+bhh)   (M=16384, N=1024, K=320)
    k_gemm<<<dim3(NNODES / 16, G4 / 512), dim3(256), 0, stream>>>(
        embX, wihF, biasF, xpF, G4, EMBP, 0);
    k_gemm<<<dim3(NNODES / 16, G4 / 512), dim3(256), 0, stream>>>(
        embX, wihB, biasB, xpB, G4, EMBP, 0);

    k_padrow<<<dim3(2), dim3(256), 0, stream>>>(padding_vec, nodeR);

    // BiLSTM recurrence (persistent, WMMA recurrent GEMM each step)
    k_lstm<<<dim3(BB / 16), dim3(256), 0, stream>>>(xpF, whhF, nodeR, out_vec, 0, 0);
    k_lstm<<<dim3(BB / 16), dim3(256), 0, stream>>>(xpB, whhB, nodeR, out_vec, 1, HHD);

    // 7 GraphSAGE layers, fw + bw  (M=16384, N=512, K=1024)
    for (int layer = 0; layer < 7; ++layer) {
        int li = layer;  // min(layer, 6)
        const float* selfF  = (layer == 0) ? nodeR : hidF;
        const float* selfBw = (layer == 0) ? nodeR : hidB;
        int padZero = (layer == 0) ? 0 : 1;
        int clen    = (layer == 0) ? 1 : 0;
        // forward
        k_gather<<<dim3(NNODES), dim3(256), 0, stream>>>(
            selfF, selfF, padZero, fw_adj, lenF, clen, concatB);
        k_gemm<<<dim3(NNODES / 16, 1), dim3(256), 0, stream>>>(
            concatB, sageWF + (size_t)li * HD * G4,
            fw_b + (size_t)li * HD, hidF, HD, G4, 1);
        // backward
        k_gather<<<dim3(NNODES), dim3(256), 0, stream>>>(
            selfBw, selfBw, padZero, bw_adj, lenB, clen, concatB);
        k_gemm<<<dim3(NNODES / 16, 1), dim3(256), 0, stream>>>(
            concatB, sageWB + (size_t)li * HD * G4,
            bw_b + (size_t)li * HD, hidB, HD, G4, 1);
    }

    // final concat + max-pool
    k_pool<<<dim3(BB), dim3(256), 0, stream>>>(hidF, hidB, out_hidden, out_gemb);
}
